// SpatioTemporal_dot_80951543595327
// MI455X (gfx1250) — matrix operand order repaired and checked by reading, not compiled, not run on hardware
//
#include <hip/hip_runtime.h>
#include <hip/hip_bf16.h>

typedef __attribute__((ext_vector_type(16))) _Float16 v16h;
typedef __attribute__((ext_vector_type(8)))  _Float16 v8h;
typedef __attribute__((ext_vector_type(2)))  __fp16   v2fp16;  // cvt_pkrtz result type
typedef __attribute__((ext_vector_type(8)))  float    v8f;
typedef __attribute__((ext_vector_type(4)))  int      v4i;

typedef __attribute__((address_space(1))) v4i* gptr_v4i;   // global int4*
typedef __attribute__((address_space(3))) v4i* lptr_v4i;   // LDS int4*

#define N_TOK   8192   // T*H*W
#define C_CH    64
#define NF      10
#define F_PAD   16
#define HW      1024
#define M_ITERS (N_TOK / 32)

// ---------------------------------------------------------------------------
// gfx1250 async global->LDS copy (ASYNCcnt-tracked, no VGPR round trip).
// Probe-confirmed: builtin takes (AS1 v4i* src, AS3 v4i* dst, int off, int cpol).
// ---------------------------------------------------------------------------
#ifdef __has_builtin
#if __has_builtin(__builtin_amdgcn_global_load_async_to_lds_b128)
#define ASYNC_LDS_BUILTIN 1
#endif
#if __has_builtin(__builtin_amdgcn_s_wait_asynccnt)
#define WAIT_ASYNC_BUILTIN 1
#endif
#endif

__device__ __forceinline__ void async_copy_b128(const _Float16* g, _Float16* l) {
#ifdef ASYNC_LDS_BUILTIN
    __builtin_amdgcn_global_load_async_to_lds_b128(
        (gptr_v4i)(uintptr_t)g, (lptr_v4i)(uintptr_t)l, 0, 0);
#else
    // VDST = LDS byte address (u32 VGPR), VADDR = 64-bit global address, GV mode.
    const unsigned int       loff  = (unsigned int)(uintptr_t)l;        // low 32 = LDS offset
    const unsigned long long gaddr = (unsigned long long)(uintptr_t)g;
    asm volatile("global_load_async_to_lds_b128 %0, %1, off"
                 :: "v"(loff), "v"(gaddr) : "memory");
#endif
}

__device__ __forceinline__ void wait_async_zero() {
#ifdef WAIT_ASYNC_BUILTIN
    __builtin_amdgcn_s_wait_asynccnt(0);
#else
    asm volatile("s_wait_asynccnt 0x0" ::: "memory");
#endif
}

// ---------------------------------------------------------------------------
// Phase 1: Q/K/V projections (~1% of total FLOPs -> scalar VALU).
//   Qh[n][16], Kh[n][16] row-major f16 (f-padded with zeros),
//   Vt[f][8192] transposed so phase-2 B-operand loads are contiguous.
// ---------------------------------------------------------------------------
__global__ __launch_bounds__(256) void proj_kernel(
    const float* __restrict__ x1, const float* __restrict__ x2,
    const float* __restrict__ w1, const float* __restrict__ b1,
    const float* __restrict__ w2, const float* __restrict__ b2,
    const float* __restrict__ w3, const float* __restrict__ b3,
    _Float16* __restrict__ Qh, _Float16* __restrict__ Kh,
    _Float16* __restrict__ Vt)
{
    const int n  = blockIdx.x * blockDim.x + threadIdx.x;   // 0..8191
    const int t  = n >> 10;
    const int hw = n & 1023;
    const float* p1 = x1 + (size_t)t * C_CH * HW + hw;
    const float* p2 = x2 + (size_t)t * C_CH * HW + hw;

    float xa[C_CH], xb[C_CH];
#pragma unroll
    for (int c = 0; c < C_CH; ++c) { xa[c] = p1[(size_t)c * HW]; xb[c] = p2[(size_t)c * HW]; }

    for (int f = 0; f < F_PAD; ++f) {
        float q = 0.f, k = 0.f, v = 0.f;
        if (f < NF) {
            q = b1[f]; k = b2[f]; v = b3[f];
#pragma unroll
            for (int c = 0; c < C_CH; ++c) {
                q = fmaf(w1[f * C_CH + c], xa[c], q);
                k = fmaf(w2[f * C_CH + c], xb[c], k);
                v = fmaf(w3[f * C_CH + c], xa[c], v);
            }
        }
        Qh[(size_t)n * F_PAD + f] = (_Float16)q;
        Kh[(size_t)n * F_PAD + f] = (_Float16)k;
        Vt[(size_t)f * N_TOK + n] = (_Float16)v;
    }
}

// ---------------------------------------------------------------------------
// Per-iteration tile staging: K tile (32x16 f16) + V^T tile (16x32 f16),
// 128 threads x one async 16B chunk each, straight into LDS.
// ---------------------------------------------------------------------------
__device__ __forceinline__ void stage_tiles(const _Float16* __restrict__ Kh,
                                            const _Float16* __restrict__ Vt,
                                            _Float16* lK, _Float16* lV,
                                            int tid, int m0)
{
    if (tid < 64) {
        const int row = tid >> 1, seg = tid & 1;
        async_copy_b128(Kh + (size_t)(m0 + row) * F_PAD + seg * 8,
                        lK + row * F_PAD + seg * 8);
        if (m0 + 64 < N_TOK)
            __builtin_prefetch(Kh + (size_t)(m0 + 64 + row) * F_PAD + seg * 8, 0, 1);
    } else {
        const int t2 = tid - 64;
        const int f = t2 >> 2, seg = t2 & 3;
        async_copy_b128(Vt + (size_t)f * N_TOK + m0 + seg * 8,
                        lV + f * 32 + seg * 8);
        if (m0 + 64 < N_TOK)
            __builtin_prefetch(Vt + (size_t)f * N_TOK + m0 + 64 + seg * 8, 0, 1);
    }
}

// ---------------------------------------------------------------------------
// Phase 2: out = relu(Q K^T) V, flash-style streaming over m.
// 4 waves/block, wave owns Q rows n0..n0+15.  Per 32-wide m-tile:
//   S(16x32) = Q * K^T   -> 2x v_wmma_f32_16x16x32_f16
//   relu (single v_max) + v_cvt_pk_rtz_f16_f32 + same-wave LDS transpose
//   O += S * V           -> 1x v_wmma_f32_16x16x32_f16
// K/V tiles double-buffered via async-to-LDS: one barrier per iteration.
// ---------------------------------------------------------------------------
__global__ __launch_bounds__(128) void attn_kernel(
    const _Float16* __restrict__ Qh, const _Float16* __restrict__ Kh,
    const _Float16* __restrict__ Vt, float* __restrict__ out)
{
    __shared__ __align__(32) _Float16 lK[2][32 * F_PAD];   // K tile   [m_local][f]
    __shared__ __align__(32) _Float16 lV[2][F_PAD * 32];   // V^T tile [f][m_local]
    __shared__ __align__(32) _Float16 sS[4][16 * 32];      // per-wave scores [n][m]

    const int tid  = threadIdx.x;
    const int lane = tid & 31;
    const int wv   = tid >> 5;
    const int hi   = lane >> 4;     // 0 = lanes 0-15, 1 = lanes 16-31
    const int l16  = lane & 15;

    const int n0 = (blockIdx.x * 4 + wv) * 16;

    // A operand (Q tile), contiguous K layout: lane<16 row l16 holds K=f{0..15},
    // lane>=16 holds K=f{16..31}.  f>=16 is zero padding.
    v16h aQ = {};
    {
        if (hi == 0) aQ = *(const v16h*)(Qh + (size_t)(n0 + l16) * F_PAD);
    }

    v8f acc = {};
    const v8f zero = {};

    stage_tiles(Kh, Vt, lK[0], lV[0], tid, 0);   // prologue: tile 0 in flight

    for (int it = 0; it < M_ITERS; ++it) {
        const int p  = it & 1;
        const int m0 = it * 32;

        wait_async_zero();   // my async writes for tile `it` are in LDS
        __syncthreads();     // everyone's writes visible; prev readers of p^1 done

        if (it + 1 < M_ITERS)
            stage_tiles(Kh, Vt, lK[p ^ 1], lV[p ^ 1], tid, m0 + 32);

        // ---- S = Q * K^T : B column n = m_local = l16, K-dim = f ----
        v16h bK1 = {}, bK2 = {};
        if (hi == 0) {
            bK1 = *(const v16h*)(lK[p] + l16 * F_PAD);          // cols m0+0..15
            bK2 = *(const v16h*)(lK[p] + (16 + l16) * F_PAD);   // cols m0+16..31
        }
        v8f s1 = __builtin_amdgcn_wmma_f32_16x16x32_f16(false, aQ, false, bK1,
                                                        (short)0, zero, false, false);
        v8f s2 = __builtin_amdgcn_wmma_f32_16x16x32_f16(false, aQ, false, bK2,
                                                        (short)0, zero, false, false);

        // ---- relu + packed f16 cvt + cross-lane transpose via per-wave LDS ----
        // (same-wave DS ops complete in order: no workgroup barrier needed)
        __fp16* s = (__fp16*)sS[wv];      // same 2-byte layout as _Float16
#pragma unroll
        for (int r = 0; r < 8; r += 2) {
            const int nl = r + 8 * hi;               // C/D layout: M = r + 8*(lane/16)
            const float a0 = s1[r]     > 0.f ? s1[r]     : 0.f;
            const float a1 = s1[r + 1] > 0.f ? s1[r + 1] : 0.f;
            const float b0 = s2[r]     > 0.f ? s2[r]     : 0.f;
            const float b1 = s2[r + 1] > 0.f ? s2[r + 1] : 0.f;
            const v2fp16 pa = __builtin_amdgcn_cvt_pkrtz(a0, a1);  // v_cvt_pk_rtz_f16_f32
            const v2fp16 pb = __builtin_amdgcn_cvt_pkrtz(b0, b1);
            s[nl * 32 + l16]            = pa.x;
            s[(nl + 1) * 32 + l16]      = pa.y;
            s[nl * 32 + 16 + l16]       = pb.x;
            s[(nl + 1) * 32 + 16 + l16] = pb.y;
        }

        // ---- A operand (S tile, 16x32): lane row n = l16, K = m_local ----
        v16h aS;
        {
            const _Float16* sr = sS[wv];
            aS = *(const v16h*)(sr + l16 * 32 + hi * 16);   // m {0..15 | 16..31}
        }
        // B operand (V tile, 32x16): lane column f = l16, K range m = hi*16+0..15
        const v16h bV = *(const v16h*)(lV[p] + l16 * 32 + hi * 16);

        acc = __builtin_amdgcn_wmma_f32_16x16x32_f16(false, aS, false, bV,
                                                     (short)0, acc, false, false);
    }

    // ---- writeback: out[t, f, h, w], n = n0 + r + 8*hi, f = l16 (< 10) ----
    if (l16 < NF) {
#pragma unroll
        for (int r = 0; r < 8; ++r) {
            const int n  = n0 + r + 8 * hi;
            const int t  = n >> 10;
            const int hw = n & 1023;
            out[(size_t)t * (NF * HW) + (size_t)l16 * HW + hw] = acc[r];
        }
    }
}

// ---------------------------------------------------------------------------
extern "C" void kernel_launch(void* const* d_in, const int* in_sizes, int n_in,
                              void* d_out, int out_size, void* d_ws, size_t ws_size,
                              hipStream_t stream) {
    const float* x1 = (const float*)d_in[0];
    const float* x2 = (const float*)d_in[1];
    const float* w1 = (const float*)d_in[2];
    const float* b1 = (const float*)d_in[3];
    const float* w2 = (const float*)d_in[4];
    const float* b2 = (const float*)d_in[5];
    const float* w3 = (const float*)d_in[6];
    const float* b3 = (const float*)d_in[7];

    // workspace: 3 x 8192x16 f16 = 768 KB
    _Float16* Qh = (_Float16*)d_ws;
    _Float16* Kh = Qh + (size_t)N_TOK * F_PAD;
    _Float16* Vt = Kh + (size_t)N_TOK * F_PAD;

    proj_kernel<<<N_TOK / 256, 256, 0, stream>>>(x1, x2, w1, b1, w2, b2, w3, b3,
                                                 Qh, Kh, Vt);
    // 512 q-tiles of 16 rows, 4 waves (tiles) per block -> 128 blocks
    attn_kernel<<<N_TOK / 64, 128, 0, stream>>>(Qh, Kh, Vt, (float*)d_out);
}
